// TransformerBlockWithLoadBalancing_76450417869445
// MI455X (gfx1250) — compile-verified, matrix operand order checked by reading, not run on hardware
//
#include <hip/hip_runtime.h>
#include <cstdint>

// ---------------------------------------------------------------------------
// Types / helpers
// ---------------------------------------------------------------------------
typedef __bf16 bf16_t;
typedef __attribute__((ext_vector_type(16))) __bf16 v16bf;
typedef __attribute__((ext_vector_type(8)))  float  v8f;

union Frag {           // one WMMA 16-bit A/B operand (8 VGPRs)
    uint4  u[2];
    v16bf  v;
};

static __device__ __forceinline__ bf16_t f2bf(float f) {
    union { float f; uint32_t u; } in;
    in.f = f;
    uint32_t r = (in.u + 0x7FFFu + ((in.u >> 16) & 1u)) >> 16;   // RNE
    union { uint16_t s; bf16_t b; } out;
    out.s = (uint16_t)r;
    return out.b;
}

static __device__ __forceinline__ void vzero8(v8f& a) {
#pragma unroll
    for (int i = 0; i < 8; ++i) a[i] = 0.0f;
}

// CDNA5 async global->LDS copy (16B per lane), tracked by ASYNCcnt.
static __device__ __forceinline__ void async_g2l_b128(void* lds, const void* g) {
    asm volatile("global_load_async_to_lds_b128 %0, %1, off"
                 :: "v"((uint32_t)(uintptr_t)lds), "v"((uint64_t)(uintptr_t)g)
                 : "memory");
}
static __device__ __forceinline__ void wait_async0() {
#if __has_builtin(__builtin_amdgcn_s_wait_asynccnt)
    __builtin_amdgcn_s_wait_asynccnt(0);
#else
    asm volatile("s_wait_asynccnt 0" ::: "memory");
#endif
}

// ---------------------------------------------------------------------------
// 0) Weight convert fp32 -> bf16, transposed to [E][N][K] (K contiguous)
// ---------------------------------------------------------------------------
__global__ void convert_transpose_kernel(const float* __restrict__ W,
                                         bf16_t* __restrict__ Wt,
                                         int Kdim, int Ndim) {
    const size_t total = (size_t)4 * Kdim * Ndim;
    for (size_t idx = (size_t)blockIdx.x * blockDim.x + threadIdx.x;
         idx < total; idx += (size_t)gridDim.x * blockDim.x) {
        size_t e   = idx / ((size_t)Kdim * Ndim);
        size_t rem = idx - e * (size_t)Kdim * Ndim;
        size_t k   = rem / Ndim;
        size_t n   = rem - k * Ndim;
        Wt[e * (size_t)Ndim * Kdim + n * Kdim + k] = f2bf(W[idx]);
    }
}

// ---------------------------------------------------------------------------
// 1) Router: probs[tok][4] = softmax(rv . rW + rb).  One wave per token.
// ---------------------------------------------------------------------------
__global__ __launch_bounds__(128) void router_kernel(
    const float* __restrict__ rv, const float* __restrict__ rW,
    const float* __restrict__ rb, float* __restrict__ probs) {
    const int lane = threadIdx.x & 31;
    const int wave = threadIdx.x >> 5;
    const int tok  = blockIdx.x * 4 + wave;
    const float* xv = rv + (size_t)tok * 1024;
    float s0 = 0.f, s1 = 0.f, s2 = 0.f, s3 = 0.f;
    for (int k = lane; k < 1024; k += 32) {
        float  x = xv[k];
        float4 w = *(const float4*)(rW + k * 4);
        s0 += x * w.x; s1 += x * w.y; s2 += x * w.z; s3 += x * w.w;
    }
#pragma unroll
    for (int off = 16; off >= 1; off >>= 1) {
        s0 += __shfl_xor(s0, off, 32);
        s1 += __shfl_xor(s1, off, 32);
        s2 += __shfl_xor(s2, off, 32);
        s3 += __shfl_xor(s3, off, 32);
    }
    if (lane == 0) {
        s0 += rb[0]; s1 += rb[1]; s2 += rb[2]; s3 += rb[3];
        float m  = fmaxf(fmaxf(s0, s1), fmaxf(s2, s3));
        float e0 = __expf(s0 - m), e1 = __expf(s1 - m);
        float e2 = __expf(s2 - m), e3 = __expf(s3 - m);
        float inv = 1.0f / (e0 + e1 + e2 + e3);
        float4 o; o.x = e0 * inv; o.y = e1 * inv; o.z = e2 * inv; o.w = e3 * inv;
        *(float4*)(probs + (size_t)tok * 4) = o;
    }
}

// ---------------------------------------------------------------------------
// 2) Soft-MoE GEMM:  Y[m][n] = sum_e p[m,e]*(X[m,:] @ We[:,n]) + sum_e p[m,e]*be[n]
//    Probs folded into A at LDS staging -> single WMMA accumulator chain.
//    Block: 256 thr, tile 128(M) x 128(N), BK=32. Wave tile 32x64 (2x4 WMMA).
//    B tile staged with async global->LDS (ASYNCcnt).  act: 0=none, 1=GELU.
// ---------------------------------------------------------------------------
#define LDA 40   // padded halves per As row (80 B, 16B-aligned rows)
#define LDB 40

__global__ __launch_bounds__(256) void moe_gemm_kernel(
    const float* __restrict__ X, const bf16_t* __restrict__ WT,
    const float* __restrict__ bias, const float* __restrict__ probs,
    float* __restrict__ Y, int Ndim, int Kdim, int act) {
    __shared__ bf16_t As[128 * LDA];
    __shared__ bf16_t Bs[128 * LDB];

    const int tid  = threadIdx.x;
    const int lane = tid & 31;
    const int wave = tid >> 5;
    const int wm   = wave >> 1;          // 0..3  (M sub-block of 32)
    const int wn   = wave & 1;           // 0..1  (N sub-block of 64)
    const int mBase = blockIdx.y * 128;
    const int nBase = blockIdx.x * 128;

    v8f acc[2][4];
#pragma unroll
    for (int i = 0; i < 2; ++i)
#pragma unroll
        for (int j = 0; j < 4; ++j) vzero8(acc[i][j]);

    // staging maps
    const int arow = tid >> 1;           // 0..127
    const int acol = (tid & 1) * 16;     // 0 or 16
    const int brow = tid >> 1;           // 0..127
    const int bcol = (tid & 1) * 16;     // 0 or 16

    for (int e = 0; e < 4; ++e) {
        const bf16_t* We = WT + (size_t)e * Ndim * Kdim;
        const float   pr = probs[(size_t)(mBase + arow) * 4 + e];
        for (int k0 = 0; k0 < Kdim; k0 += 32) {
            __syncthreads();
            {   // stage B via async DMA: 128 rows x 32 halves (2x16B per lane)
                const bf16_t* wp = We + (size_t)(nBase + brow) * Kdim + k0 + bcol;
                bf16_t* dst = Bs + brow * LDB + bcol;
                async_g2l_b128(dst,     wp);
                async_g2l_b128(dst + 8, wp + 8);
            }
            {   // stage A (prob-scaled, fp32 -> bf16) on the VALU path
                const float* xp = X + (size_t)(mBase + arow) * Kdim + k0 + acol;
                if (k0 + 32 < Kdim) __builtin_prefetch(xp + 32, 0, 1);
                bf16_t* dst = As + arow * LDA + acol;
#pragma unroll
                for (int q = 0; q < 4; ++q) {
                    float4 f = *(const float4*)(xp + q * 4);
                    dst[q * 4 + 0] = f2bf(f.x * pr);
                    dst[q * 4 + 1] = f2bf(f.y * pr);
                    dst[q * 4 + 2] = f2bf(f.z * pr);
                    dst[q * 4 + 3] = f2bf(f.w * pr);
                }
            }
            wait_async0();                 // our async B chunks are in LDS
            __syncthreads();               // -> whole tile visible to all waves

            const int mrow = lane & 15;
            const int hi   = lane >> 4;
            Frag a[2], b[4];
#pragma unroll
            for (int mt = 0; mt < 2; ++mt) {
                const bf16_t* p = As + (wm * 32 + mt * 16 + mrow) * LDA;
                a[mt].u[0] = *(const uint4*)(p + hi * 8);
                a[mt].u[1] = *(const uint4*)(p + 16 + hi * 8);
            }
#pragma unroll
            for (int nt = 0; nt < 4; ++nt) {
                const bf16_t* p = Bs + (wn * 64 + nt * 16 + mrow) * LDB;
                b[nt].u[0] = *(const uint4*)(p + hi * 8);
                b[nt].u[1] = *(const uint4*)(p + 16 + hi * 8);
            }
#pragma unroll
            for (int mt = 0; mt < 2; ++mt)
#pragma unroll
                for (int nt = 0; nt < 4; ++nt)
                    acc[mt][nt] = __builtin_amdgcn_wmma_f32_16x16x32_bf16(
                        false, a[mt].v, false, b[nt].v,
                        (short)0, acc[mt][nt], false, false);
        }
    }

    // epilogue: + probs @ bias, optional GELU
    const int hi2  = lane >> 4;
    const int ln15 = lane & 15;
#pragma unroll
    for (int nt = 0; nt < 4; ++nt) {
        const int col = nBase + wn * 64 + nt * 16 + ln15;
        const float bc0 = bias[0 * (size_t)Ndim + col];
        const float bc1 = bias[1 * (size_t)Ndim + col];
        const float bc2 = bias[2 * (size_t)Ndim + col];
        const float bc3 = bias[3 * (size_t)Ndim + col];
#pragma unroll
        for (int mt = 0; mt < 2; ++mt) {
#pragma unroll
            for (int r = 0; r < 8; ++r) {
                const int row = mBase + wm * 32 + mt * 16 + r + hi2 * 8;
                const float4 p = *(const float4*)(probs + (size_t)row * 4);
                float v = acc[mt][nt][r] + p.x * bc0 + p.y * bc1 + p.z * bc2 + p.w * bc3;
                if (act) v = 0.5f * v * (1.0f + erff(v * 0.70710678118654752f));
                Y[(size_t)row * Ndim + col] = v;
            }
        }
    }
}

// ---------------------------------------------------------------------------
// 3) QKV split + RoPE.  q scaled by 1/sqrt(HD)=0.125 here.
//    Outputs: qb,kb bf16 [B*H][S][64]; vT bf16 [B*H][64][S] (for O^T frag).
// ---------------------------------------------------------------------------
__global__ void rope_split_kernel(const float* __restrict__ qkv,
                                  bf16_t* __restrict__ qb,
                                  bf16_t* __restrict__ kb,
                                  bf16_t* __restrict__ vT) {
    const int idx = blockIdx.x * blockDim.x + threadIdx.x;  // 4096*16*32 threads
    const int j   = idx & 31;
    const int h   = (idx >> 5) & 15;
    const int tok = idx >> 9;
    if (tok >= 4096) return;
    const int b = tok >> 11, s = tok & 2047;

    const float inv_freq = __powf(10000.0f, -((float)j) / 32.0f);  // 2j/HD
    float sn, cs;
    __sincosf((float)s * inv_freq, &sn, &cs);

    const float* base = qkv + (size_t)tok * 3072 + h * 64 + j;
    const float q1 = base[0],         q2 = base[32];
    const float k1 = base[1024],      k2 = base[1024 + 32];
    const float v1 = base[2048],      v2 = base[2048 + 32];

    const size_t qo = ((size_t)(b * 16 + h) * 2048 + s) * 64 + j;
    qb[qo]      = f2bf((q1 * cs - q2 * sn) * 0.125f);
    qb[qo + 32] = f2bf((q2 * cs + q1 * sn) * 0.125f);
    kb[qo]      = f2bf(k1 * cs - k2 * sn);
    kb[qo + 32] = f2bf(k2 * cs + k1 * sn);

    const size_t vo = ((size_t)(b * 16 + h) * 64 + j) * 2048 + s;
    vT[vo]               = f2bf(v1);
    vT[vo + 32 * 2048]   = f2bf(v2);
}

// ---------------------------------------------------------------------------
// 4) Causal flash attention, fully transposed (S^T = K Q^T, O^T = V^T P^T).
//    Block 128 thr (4 waves), each wave owns 16 q rows; 64-q block per WG.
//    KV tiles of 32 staged via async global->LDS, shared by all 4 waves.
// ---------------------------------------------------------------------------
#define LDK 72   // 144 B row stride (16B-aligned)
#define LDV 40   // 80 B row stride

__global__ __launch_bounds__(128) void attention_kernel(
    const bf16_t* __restrict__ qg, const bf16_t* __restrict__ kg,
    const bf16_t* __restrict__ vtg, float* __restrict__ out) {
    __shared__ bf16_t Ks[32 * LDK];
    __shared__ bf16_t Vs[64 * LDV];

    const int tid  = threadIdx.x;
    const int lane = tid & 31;
    const int wave = tid >> 5;
    const int bh   = blockIdx.y;                 // 0..31
    const int qb0  = blockIdx.x * 64;
    const int hi   = lane >> 4;
    const int qrow = qb0 + wave * 16 + (lane & 15);   // this lane's q row

    // Q fragments (B-operand role): lane n = q-row, halves = d chunks
    Frag qf[2];
    {
        const bf16_t* qp = qg + ((size_t)bh * 2048 + qrow) * 64;
        qf[0].u[0] = *(const uint4*)(qp + hi * 8);
        qf[0].u[1] = *(const uint4*)(qp + 16 + hi * 8);
        qf[1].u[0] = *(const uint4*)(qp + 32 + hi * 8);
        qf[1].u[1] = *(const uint4*)(qp + 48 + hi * 8);
    }

    v8f ot[4];
#pragma unroll
    for (int t = 0; t < 4; ++t) vzero8(ot[t]);
    float m_run = -3.0e38f, l_run = 0.0f;

    const int kvEnd = qb0 + 64;
    for (int kv0 = 0; kv0 < kvEnd; kv0 += 32) {
        __syncthreads();
        {   // stage K tile: 32 rows x 64 halves (async DMA, 2x16B per lane)
            const int r = tid >> 2, seg = (tid & 3) * 16;
            const bf16_t* src = kg + ((size_t)bh * 2048 + kv0 + r) * 64 + seg;
            bf16_t* dst = Ks + r * LDK + seg;
            async_g2l_b128(dst,     src);
            async_g2l_b128(dst + 8, src + 8);
        }
        {   // stage V^T tile: 64 rows(hd) x 32 halves(kv) (async DMA)
            const int r = tid >> 1, seg = (tid & 1) * 16;
            const bf16_t* src = vtg + ((size_t)bh * 64 + r) * 2048 + kv0 + seg;
            bf16_t* dst = Vs + r * LDV + seg;
            async_g2l_b128(dst,     src);
            async_g2l_b128(dst + 8, src + 8);
        }
        wait_async0();
        __syncthreads();

        // S^T tiles: st[t] is 16(kv) x 16(q)
        v8f st[2];
#pragma unroll
        for (int t = 0; t < 2; ++t) {
            vzero8(st[t]);
            Frag ka0, ka1;
            const bf16_t* kp = Ks + (t * 16 + (lane & 15)) * LDK;
            ka0.u[0] = *(const uint4*)(kp + hi * 8);
            ka0.u[1] = *(const uint4*)(kp + 16 + hi * 8);
            ka1.u[0] = *(const uint4*)(kp + 32 + hi * 8);
            ka1.u[1] = *(const uint4*)(kp + 48 + hi * 8);
            st[t] = __builtin_amdgcn_wmma_f32_16x16x32_bf16(
                false, ka0.v, false, qf[0].v, (short)0, st[t], false, false);
            st[t] = __builtin_amdgcn_wmma_f32_16x16x32_bf16(
                false, ka1.v, false, qf[1].v, (short)0, st[t], false, false);
        }

        // causal mask + tile max (per-lane: one q row)
        float tmax = -3.0e38f;
#pragma unroll
        for (int t = 0; t < 2; ++t)
#pragma unroll
            for (int r = 0; r < 8; ++r) {
                const int kvg = kv0 + t * 16 + r + hi * 8;
                float v = st[t][r];
                if (kvg > qrow) v = -3.0e38f;
                st[t][r] = v;
                tmax = fmaxf(tmax, v);
            }
        tmax = fmaxf(tmax, __shfl_xor(tmax, 16, 32));
        const float m_new  = fmaxf(m_run, tmax);
        const float escale = __expf(m_run - m_new);

        // P (bf16) packed straight into the P^T B-fragment of this lane
        Frag pf;
        float rs = 0.0f;
#pragma unroll
        for (int r = 0; r < 8; ++r) {
            const float p0 = __expf(st[0][r] - m_new);
            const float p1 = __expf(st[1][r] - m_new);
            rs += p0 + p1;
            pf.v[r]     = f2bf(p0);
            pf.v[r + 8] = f2bf(p1);
        }
        rs += __shfl_xor(rs, 16, 32);
        l_run = l_run * escale + rs;
        m_run = m_new;

#pragma unroll
        for (int t = 0; t < 4; ++t)
#pragma unroll
            for (int r = 0; r < 8; ++r) ot[t][r] *= escale;

        // O^T += V^T(16x32) x P^T(32x16) per hd tile
#pragma unroll
        for (int hd = 0; hd < 4; ++hd) {
            Frag vf;
            const bf16_t* vp = Vs + (hd * 16 + (lane & 15)) * LDV;
            vf.u[0] = *(const uint4*)(vp + hi * 8);
            vf.u[1] = *(const uint4*)(vp + 16 + hi * 8);
            ot[hd] = __builtin_amdgcn_wmma_f32_16x16x32_bf16(
                false, vf.v, false, pf.v, (short)0, ot[hd], false, false);
        }
    }

    const float invl = 1.0f / l_run;
    const int   tok  = (bh >> 4) * 2048 + qrow;
#pragma unroll
    for (int hd = 0; hd < 4; ++hd)
#pragma unroll
        for (int r = 0; r < 8; ++r) {
            const int d = (bh & 15) * 64 + hd * 16 + r + hi * 8;
            out[(size_t)tok * 1024 + d] = ot[hd][r] * invl;
        }
}

// ---------------------------------------------------------------------------
// 5) out = LayerNorm(x + y) over D=1024.  One block (256 thr) per token.
// ---------------------------------------------------------------------------
__global__ __launch_bounds__(256) void add_ln_kernel(
    const float* __restrict__ x, const float* __restrict__ y,
    const float* __restrict__ g, const float* __restrict__ bb,
    float* __restrict__ out) {
    __shared__ float red[8];
    const int tok = blockIdx.x;
    const float* xp = x + (size_t)tok * 1024;
    const float* yp = y + (size_t)tok * 1024;
    float v[4];
    float s = 0.0f;
#pragma unroll
    for (int i = 0; i < 4; ++i) {
        const int d = threadIdx.x + i * 256;
        v[i] = xp[d] + yp[d];
        s += v[i];
    }
#pragma unroll
    for (int off = 16; off >= 1; off >>= 1) s += __shfl_xor(s, off, 32);
    if ((threadIdx.x & 31) == 0) red[threadIdx.x >> 5] = s;
    __syncthreads();
    float tot = 0.0f;
#pragma unroll
    for (int i = 0; i < 8; ++i) tot += red[i];
    const float mu = tot * (1.0f / 1024.0f);
    __syncthreads();

    float s2 = 0.0f;
#pragma unroll
    for (int i = 0; i < 4; ++i) { const float d = v[i] - mu; s2 += d * d; }
#pragma unroll
    for (int off = 16; off >= 1; off >>= 1) s2 += __shfl_xor(s2, off, 32);
    if ((threadIdx.x & 31) == 0) red[threadIdx.x >> 5] = s2;
    __syncthreads();
    float tot2 = 0.0f;
#pragma unroll
    for (int i = 0; i < 8; ++i) tot2 += red[i];
    const float rstd = rsqrtf(tot2 * (1.0f / 1024.0f) + 1e-5f);
#pragma unroll
    for (int i = 0; i < 4; ++i) {
        const int d = threadIdx.x + i * 256;
        out[(size_t)tok * 1024 + d] = (v[i] - mu) * rstd * g[d] + bb[d];
    }
}

// ---------------------------------------------------------------------------
// Launch
// ---------------------------------------------------------------------------
extern "C" void kernel_launch(void* const* d_in, const int* in_sizes, int n_in,
                              void* d_out, int out_size, void* d_ws, size_t ws_size,
                              hipStream_t stream) {
    const float* x    = (const float*)d_in[0];
    const float* rv   = (const float*)d_in[1];
    const float* rW   = (const float*)d_in[2];
    const float* rb   = (const float*)d_in[3];
    const float* qkvW = (const float*)d_in[4];
    const float* qkvB = (const float*)d_in[5];
    const float* oW   = (const float*)d_in[6];
    const float* oB   = (const float*)d_in[7];
    const float* miW  = (const float*)d_in[8];
    const float* miB  = (const float*)d_in[9];
    const float* moW  = (const float*)d_in[10];
    const float* moB  = (const float*)d_in[11];
    const float* ln1g = (const float*)d_in[12];
    const float* ln1b = (const float*)d_in[13];
    const float* ln2g = (const float*)d_in[14];
    const float* ln2b = (const float*)d_in[15];
    float* out = (float*)d_out;

    char* ws = (char*)d_ws;
    size_t off = 0;
    auto alloc = [&](size_t bytes) -> void* {
        void* p = ws + off;
        off = (off + bytes + 255) & ~(size_t)255;
        return p;
    };
    float* probs = (float*)alloc((size_t)4096 * 4 * 4);
    float* qkv   = (float*)alloc((size_t)4096 * 3072 * 4);
    float* attn  = (float*)alloc((size_t)4096 * 1024 * 4);
    float* x2    = (float*)alloc((size_t)4096 * 1024 * 4);
    float* hbuf  = (float*)alloc((size_t)4096 * 4096 * 4);
    float* tmp   = (float*)alloc((size_t)4096 * 1024 * 4);
    bf16_t* qb   = (bf16_t*)alloc((size_t)32 * 2048 * 64 * 2);
    bf16_t* kb   = (bf16_t*)alloc((size_t)32 * 2048 * 64 * 2);
    bf16_t* vT   = (bf16_t*)alloc((size_t)32 * 2048 * 64 * 2);
    bf16_t* qkvT = (bf16_t*)alloc((size_t)4 * 3072 * 1024 * 2);
    bf16_t* oT   = (bf16_t*)alloc((size_t)4 * 1024 * 1024 * 2);
    bf16_t* miT  = (bf16_t*)alloc((size_t)4 * 4096 * 1024 * 2);
    bf16_t* moT  = (bf16_t*)alloc((size_t)4 * 1024 * 4096 * 2);
    (void)in_sizes; (void)n_in; (void)out_size; (void)ws_size;

    // weights -> bf16, [E][N][K]
    convert_transpose_kernel<<<4096, 256, 0, stream>>>(qkvW, qkvT, 1024, 3072);
    convert_transpose_kernel<<<4096, 256, 0, stream>>>(oW,   oT,   1024, 1024);
    convert_transpose_kernel<<<4096, 256, 0, stream>>>(miW,  miT,  1024, 4096);
    convert_transpose_kernel<<<4096, 256, 0, stream>>>(moW,  moT,  4096, 1024);

    router_kernel<<<1024, 128, 0, stream>>>(rv, rW, rb, probs);

    // qkv = moe(x)  [4096 x 3072]
    moe_gemm_kernel<<<dim3(24, 32), 256, 0, stream>>>(x, qkvT, qkvB, probs, qkv, 3072, 1024, 0);

    rope_split_kernel<<<(4096 * 16 * 32) / 256, 256, 0, stream>>>(qkv, qb, kb, vT);

    attention_kernel<<<dim3(32, 32), 128, 0, stream>>>(qb, kb, vT, attn);

    // o-proj: tmp = moe(attn)  [4096 x 1024]
    moe_gemm_kernel<<<dim3(8, 32), 256, 0, stream>>>(attn, oT, oB, probs, tmp, 1024, 1024, 0);
    add_ln_kernel<<<4096, 256, 0, stream>>>(x, tmp, ln1g, ln1b, x2);

    // mlp_in + GELU: hbuf = gelu(moe(x2))  [4096 x 4096]
    moe_gemm_kernel<<<dim3(32, 32), 256, 0, stream>>>(x2, miT, miB, probs, hbuf, 4096, 1024, 1);
    // mlp_out: tmp = moe(hbuf)  [4096 x 1024]
    moe_gemm_kernel<<<dim3(8, 32), 256, 0, stream>>>(hbuf, moT, moB, probs, tmp, 1024, 4096, 0);

    add_ln_kernel<<<4096, 256, 0, stream>>>(x2, tmp, ln2g, ln2b, out);
}
